// Attention_74491912782063
// MI455X (gfx1250) — matrix operand order repaired and checked by reading, not compiled, not run on hardware
//
#include <hip/hip_runtime.h>
#include <hip/hip_fp16.h>

typedef __attribute__((ext_vector_type(16))) _Float16 v16h;
typedef __attribute__((ext_vector_type(8)))  float    v8f;
typedef __attribute__((ext_vector_type(4)))  unsigned u32x4;
typedef __attribute__((ext_vector_type(4)))  int      i32x4;
typedef __attribute__((ext_vector_type(8)))  int      i32x8;

constexpr int kEMB  = 768;
constexpr int kNH   = 12;
constexpr int kHD   = 64;
constexpr int kSEQ  = 2048;
constexpr int kB    = 4;
constexpr int kM    = kB * kSEQ;      // 8192 rows

#if defined(__has_builtin)
#if __has_builtin(__builtin_amdgcn_tensor_load_to_lds)
#define USE_TDM 1
#else
#define USE_TDM 0
#endif
#else
#define USE_TDM 0
#endif

union FragAB { v16h h; unsigned u[8]; };

__device__ __forceinline__ v8f wmma_f16(const FragAB& a, const FragAB& b, v8f c) {
  return __builtin_amdgcn_wmma_f32_16x16x32_f16(false, a.h, false, b.h,
                                                (short)0, c, false, false);
}

// A fragment (16-bit A 16x32): lane = row, K-pairs per VGPR; rowp -> 32 halfs.
__device__ __forceinline__ void load_a_frag(FragAB& a, const _Float16* rowp, int lh) {
  const unsigned* p = (const unsigned*)rowp;
#pragma unroll
  for (int v = 0; v < 4; ++v) a.u[v]     = p[lh * 4 + v];
#pragma unroll
  for (int v = 0; v < 4; ++v) a.u[4 + v] = p[8 + lh * 4 + v];
}

// B fragment (16-bit B 32x16) from K-major storage: rowp points at 32 halfs of
// the N-column's K run; same K order as A: VGPR v<4 holds K = lh*8 + {2v, 2v+1}, v>=4 holds K = 16 + lh*8 + {2(v-4), 2(v-4)+1}.
__device__ __forceinline__ void load_b_frag(FragAB& b, const _Float16* rowp, int lh) {
  const unsigned* p = (const unsigned*)rowp;
#pragma unroll
  for (int v = 0; v < 4; ++v) b.u[v]     = p[lh * 4 + v];
#pragma unroll
  for (int v = 0; v < 4; ++v) b.u[4 + v] = p[8 + lh * 4 + v];
}

__device__ __forceinline__ unsigned lds_byte_off(const void* p) {
  // Flat shared aperture keeps the LDS offset in addr[31:0].
  return (unsigned)(unsigned long long)p;
}

#if USE_TDM
// TDM: DMA a 64x64 f16 tile (row pitch = pitch_elems) from global into LDS,
// padding each 128B row with 8B (-> matches the [64][68] padded LDS layout).
__device__ __forceinline__ void tdm_load_tile64(unsigned lds_off, const void* gptr,
                                                unsigned pitch_elems) {
  unsigned long long ga = (unsigned long long)gptr;
  u32x4 g0;
  g0.x = 1u;                                            // count=1, user D#
  g0.y = lds_off;                                       // lds_addr (bytes)
  g0.z = (unsigned)ga;                                  // global_addr[31:0]
  g0.w = (unsigned)((ga >> 32) & 0x1FFFFFFu) | (2u << 30);  // addr[56:32] | type=2
  i32x8 g1;
  // data_size=2B | pad_enable | pad_interval=32dw (code 4) | pad_amount=2dw (code 1)
  g1[0] = (int)((1u << 16) | (1u << 20) | (4u << 22) | (1u << 25));
  g1[1] = (int)(64u << 16);            // tensor_dim0 = 64  (bits 79:48, lo16)
  g1[2] = (int)(64u << 16);            // tensor_dim0 hi=0 | tensor_dim1 = 64
  g1[3] = (int)(64u << 16);            // tensor_dim1 hi=0 | tile_dim0 = 64
  g1[4] = 64;                          // tile_dim1 = 64, tile_dim2 = 0
  g1[5] = (int)pitch_elems;            // tensor_dim0_stride (data_size units)
  g1[6] = 0;
  g1[7] = 0;
  i32x4 z4 = {0, 0, 0, 0};
#if __clang_major__ >= 23
  i32x8 z8 = {0, 0, 0, 0, 0, 0, 0, 0};
  __builtin_amdgcn_tensor_load_to_lds(g0, g1, z4, z4, z8, 0);
#else
  __builtin_amdgcn_tensor_load_to_lds(g0, g1, z4, z4, 0);
#endif
}
#endif

// ---------------------------------------------------------------------------
// Mask bit-pack: one u32 per 32 consecutive mask ints via wave32 ballot.
// mask is [B,1,S,S] int32; bits is [B*S, S/32] u32.
// ---------------------------------------------------------------------------
__global__ __launch_bounds__(256)
void pack_mask_kernel(const int* __restrict__ mask, unsigned* __restrict__ bits) {
  size_t i = (size_t)blockIdx.x * 256 + threadIdx.x;
  int mv = mask[i];
  unsigned long long bal = __ballot(mv != 0);
  if ((threadIdx.x & 31) == 0) bits[i >> 5] = (unsigned)bal;
}

// ---------------------------------------------------------------------------
// Tiled GEMM: C[M x 768] = A[M x 768] * W[768 x 768]
// IN_F16: A f16 row-major, else f32 (converted in LDS staging).
// OUTMODE: 0 = f32 row-major, 1 = f16 [B,H,S,64], 2 = f16 [B,H,64,S] (V^T).
// 128 threads = 4 waves; tile 64x64x32; W tile stored K-major in LDS.
// ---------------------------------------------------------------------------
template <bool IN_F16, int OUTMODE>
__global__ __launch_bounds__(128)
void gemm768_kernel(const void* __restrict__ Aptr,
                    const float* __restrict__ W,
                    void* __restrict__ Out) {
  __shared__ _Float16 As[64][32];
  __shared__ _Float16 BsT[64][34];     // [n][k], +2 halfs pad per row

  const int tid  = threadIdx.x;
  const int wave = tid >> 5;
  const int lane = tid & 31;
  const int l16  = lane & 15;
  const int lh   = lane >> 4;
  const int row0 = blockIdx.x * 64;
  const int col0 = blockIdx.y * 64;

  v8f acc[4] = {};

  for (int k0 = 0; k0 < kEMB; k0 += 32) {
    __syncthreads();
    for (int i = tid; i < 64 * 32; i += 128) {
      int r = i >> 5, c = i & 31;
      if (IN_F16)
        As[r][c] = ((const _Float16*)Aptr)[(size_t)(row0 + r) * kEMB + k0 + c];
      else
        As[r][c] = (_Float16)((const float*)Aptr)[(size_t)(row0 + r) * kEMB + k0 + c];
    }
    for (int i = tid; i < 32 * 64; i += 128) {
      int r = i >> 6, c = i & 63;     // coalesced W read, transposed LDS store
      BsT[c][r] = (_Float16)W[(size_t)(k0 + r) * kEMB + col0 + c];
    }
    __syncthreads();

    if (k0 + 32 < kEMB)  // prefetch next W tile -> global_prefetch_b8
      __builtin_prefetch(&W[(size_t)(k0 + 32 + (tid >> 2)) * kEMB + col0 + (tid & 3) * 16], 0, 0);

    FragAB a;
    load_a_frag(a, &As[wave * 16 + l16][0], lh);
#pragma unroll
    for (int t = 0; t < 4; ++t) {
      FragAB bf;
      load_b_frag(bf, &BsT[t * 16 + l16][0], lh);
      acc[t] = wmma_f16(a, bf, acc[t]);
    }
  }

#pragma unroll
  for (int r = 0; r < 8; ++r) {
    int mrow = row0 + wave * 16 + r + lh * 8;
#pragma unroll
    for (int t = 0; t < 4; ++t) {
      int ncol = col0 + t * 16 + l16;
      if (OUTMODE == 0) {
        ((float*)Out)[(size_t)mrow * kEMB + ncol] = acc[t][r];
      } else {
        int bb = mrow >> 11, ss = mrow & (kSEQ - 1);
        int hh = ncol >> 6,  dd = ncol & (kHD - 1);
        size_t idx = (OUTMODE == 1)
            ? (((size_t)bb * kNH + hh) * kSEQ + ss) * kHD + dd
            : (((size_t)bb * kNH + hh) * kHD + dd) * kSEQ + ss;
        ((_Float16*)Out)[idx] = (_Float16)acc[t][r];
      }
    }
  }
}

// ---------------------------------------------------------------------------
// Flash-attention core. q16/k16 are [B,H,S,64] f16; v16T is [B,H,64,S] f16.
// mbits is the packed mask ([B*S, S/32] u32). One WG per (b, h, 64 q rows);
// K/V tiles double-buffered via TDM DMA, tracked with TENSORcnt.
// ---------------------------------------------------------------------------
__global__ __launch_bounds__(128)
void attn_flash_kernel(const _Float16* __restrict__ q16,
                       const _Float16* __restrict__ k16,
                       const _Float16* __restrict__ v16T,
                       const unsigned* __restrict__ mbits,
                       _Float16* __restrict__ attn_out) {
  __shared__ _Float16 qs[64][68];
  __shared__ _Float16 ks2[2][64][68];     // [key][d]
  __shared__ _Float16 vs2[2][64][68];     // [d][key]  (K-major for B frags)
  __shared__ _Float16 ps[4][16][68];

  const int tid  = threadIdx.x;
  const int wave = tid >> 5;
  const int lane = tid & 31;
  const int l16  = lane & 15;
  const int lh   = lane >> 4;
  const int qb   = blockIdx.x;
  const int h    = blockIdx.y;
  const int b    = blockIdx.z;

  const size_t headoff = ((size_t)b * kNH + h) * kSEQ * kHD;  // same for V^T
  const int q0 = qb * 64;

#if USE_TDM
  tdm_load_tile64(lds_byte_off(&qs[0][0]),     q16  + headoff + (size_t)q0 * kHD, kHD);
  tdm_load_tile64(lds_byte_off(&ks2[0][0][0]), k16  + headoff,                    kHD);
  tdm_load_tile64(lds_byte_off(&vs2[0][0][0]), v16T + headoff,                    kSEQ);
#else
  for (int i = tid; i < 64 * 64; i += 128) {
    int r = i >> 6, c = i & 63;
    qs[r][c] = q16[headoff + (size_t)(q0 + r) * kHD + c];
  }
#endif

  v8f  o[4] = {};
  float mrow[8], lrow[8];
#pragma unroll
  for (int r = 0; r < 8; ++r) { mrow[r] = -3.0e38f; lrow[r] = 0.0f; }

  const float scale = 0.036084391824351615f;   // 1/sqrt(768) (embed-dim scale)

  for (int kb = 0; kb < kSEQ / 64; ++kb) {
    const int cur = kb & 1;
#if USE_TDM
    __builtin_amdgcn_s_wait_tensorcnt(0);
    __syncthreads();
    if (kb + 1 < kSEQ / 64) {
      const int nxt = (kb + 1) & 1;
      tdm_load_tile64(lds_byte_off(&ks2[nxt][0][0]),
                      k16  + headoff + (size_t)(kb + 1) * 64 * kHD, kHD);
      tdm_load_tile64(lds_byte_off(&vs2[nxt][0][0]),
                      v16T + headoff + (size_t)(kb + 1) * 64, kSEQ);
    }
#else
    __syncthreads();
    for (int i = tid; i < 64 * 64; i += 128) {
      int r = i >> 6, c = i & 63;
      ks2[cur][r][c] = k16[headoff + (size_t)(kb * 64 + r) * kHD + c];
      vs2[cur][r][c] = v16T[headoff + (size_t)r * kSEQ + kb * 64 + c];
    }
    __syncthreads();
#endif

    // ---- energy = q @ k^T : [16 q-rows x 64 keys] per wave ----
    v8f e[4] = {};
#pragma unroll
    for (int kh = 0; kh < 2; ++kh) {
      FragAB a;
      load_a_frag(a, &qs[wave * 16 + l16][kh * 32], lh);
#pragma unroll
      for (int t = 0; t < 4; ++t) {
        FragAB bf;
        load_b_frag(bf, &ks2[cur][t * 16 + l16][kh * 32], lh);
        e[t] = wmma_f16(a, bf, e[t]);
      }
    }

    // ---- mask (packed bits, one b64 per row) + online softmax ----
    float p[4][8];
#pragma unroll
    for (int r = 0; r < 8; ++r) {
      int qrow = q0 + wave * 16 + r + lh * 8;
      unsigned long long mk = *(const unsigned long long*)
          &mbits[((size_t)b * kSEQ + qrow) * (kSEQ / 32) + kb * 2];
      float best = -3.0e38f;
#pragma unroll
      for (int t = 0; t < 4; ++t) {
        int bit = t * 16 + l16;
        float ev = ((mk >> bit) & 1ull) ? e[t][r] * scale : -3.6e18f;
        p[t][r] = ev;
        best = fmaxf(best, ev);
      }
      best = fmaxf(best, __shfl_xor(best, 1));
      best = fmaxf(best, __shfl_xor(best, 2));
      best = fmaxf(best, __shfl_xor(best, 4));
      best = fmaxf(best, __shfl_xor(best, 8));
      float mnew  = fmaxf(mrow[r], best);
      float alpha = __expf(mrow[r] - mnew);
      float rs = 0.0f;
#pragma unroll
      for (int t = 0; t < 4; ++t) {
        float pe = __expf(p[t][r] - mnew);
        p[t][r] = pe;
        rs += pe;
      }
      rs += __shfl_xor(rs, 1);
      rs += __shfl_xor(rs, 2);
      rs += __shfl_xor(rs, 4);
      rs += __shfl_xor(rs, 8);
      lrow[r] = lrow[r] * alpha + rs;
      mrow[r] = mnew;
#pragma unroll
      for (int t = 0; t < 4; ++t) o[t][r] *= alpha;
    }

    // ---- P (D-layout) -> per-wave LDS -> A-layout ----
#pragma unroll
    for (int t = 0; t < 4; ++t)
#pragma unroll
      for (int r = 0; r < 8; ++r)
        ps[wave][r + lh * 8][t * 16 + l16] = (_Float16)p[t][r];

    asm volatile("s_wait_dscnt 0x0" ::: "memory");  // wave-local LDS RAW fence

    // ---- O += P @ V ----
#pragma unroll
    for (int kh = 0; kh < 2; ++kh) {
      FragAB a;
      load_a_frag(a, &ps[wave][l16][kh * 32], lh);
#pragma unroll
      for (int t = 0; t < 4; ++t) {
        FragAB bf;
        load_b_frag(bf, &vs2[cur][t * 16 + l16][kh * 32], lh);
        o[t] = wmma_f16(a, bf, o[t]);
      }
    }
  }

  // ---- normalize, store [B*S, 768] f16 row-major ----
#pragma unroll
  for (int r = 0; r < 8; ++r) {
    float inv = 1.0f / lrow[r];
    int qrow = q0 + wave * 16 + r + lh * 8;
#pragma unroll
    for (int t = 0; t < 4; ++t) {
      int d = t * 16 + l16;
      attn_out[((size_t)(b * kSEQ) + qrow) * kEMB + h * kHD + d] =
          (_Float16)(o[t][r] * inv);
    }
  }
}

// ---------------------------------------------------------------------------
extern "C" void kernel_launch(void* const* d_in, const int* in_sizes, int n_in,
                              void* d_out, int out_size, void* d_ws, size_t ws_size,
                              hipStream_t stream) {
  (void)in_sizes; (void)n_in; (void)out_size; (void)ws_size;
  const float* Q    = (const float*)d_in[0];
  const float* K    = (const float*)d_in[1];
  const float* V    = (const float*)d_in[2];
  const int*   mask = (const int*)d_in[3];
  const float* Wq   = (const float*)d_in[4];
  const float* Wk   = (const float*)d_in[5];
  const float* Wv   = (const float*)d_in[6];
  const float* Wfc  = (const float*)d_in[7];
  float* out = (float*)d_out;

  const size_t elems = (size_t)kM * kEMB;
  _Float16* q16  = (_Float16*)d_ws;
  _Float16* k16  = q16 + elems;
  _Float16* v16T = k16 + elems;
  _Float16* ao   = v16T + elems;
  unsigned* mb   = (unsigned*)(ao + elems);      // B*S*S/32 u32 = 2 MB

  dim3 blk(128);
  dim3 gG(kM / 64, kEMB / 64);                   // 128 x 12

  pack_mask_kernel<<<(kB * kSEQ * kSEQ) / 256, 256, 0, stream>>>(mask, mb);

  gemm768_kernel<false, 1><<<gG, blk, 0, stream>>>(Q, Wq, q16);
  gemm768_kernel<false, 1><<<gG, blk, 0, stream>>>(K, Wk, k16);
  gemm768_kernel<false, 2><<<gG, blk, 0, stream>>>(V, Wv, v16T);

  dim3 gA(kSEQ / 64, kNH, kB);                   // 32 x 12 x 4
  attn_flash_kernel<<<gA, blk, 0, stream>>>(q16, k16, v16T, mb, ao);

  gemm768_kernel<true, 0><<<gG, blk, 0, stream>>>(ao, Wfc, out);
}
